// HC2WideDeepModel_86955907875178
// MI455X (gfx1250) — compile-verified, matrix operand order checked by reading, not run on hardware
//
#include <hip/hip_runtime.h>
#include <hip/hip_bf16.h>

// ---------------------------------------------------------------------------
// Types for CDNA5 WMMA (wave32): v_wmma_f32_16x16x32_bf16
// ---------------------------------------------------------------------------
typedef __attribute__((ext_vector_type(16))) __bf16 v16bf;
typedef __attribute__((ext_vector_type(8)))  __bf16 v8bf;
typedef __attribute__((ext_vector_type(8)))  float  v8f;
typedef __attribute__((ext_vector_type(4)))  int    v4i;

__device__ __forceinline__ unsigned short f32_to_bf16(float f) {
    unsigned int u = __float_as_uint(f);
    unsigned int r = u + 0x7FFFu + ((u >> 16) & 1u);   // round-to-nearest-even
    return (unsigned short)(r >> 16);
}

// --- CDNA5 async global->LDS path (ASYNCcnt), with graceful fallback --------
#if __has_builtin(__builtin_amdgcn_global_load_async_to_lds_b128)
#define HAVE_ASYNC_LDS 1
#define GL_ASYNC_B128(SRC, DST)                                              \
    __builtin_amdgcn_global_load_async_to_lds_b128(                          \
        (__attribute__((address_space(1))) v4i*)(SRC),                       \
        (__attribute__((address_space(3))) v4i*)(DST), 0, 0)
#else
#define HAVE_ASYNC_LDS 0
#endif

#if __has_builtin(__builtin_amdgcn_s_wait_asynccnt)
#define WAIT_ASYNC(n) __builtin_amdgcn_s_wait_asynccnt(n)
#else
#define WAIT_ASYNC(n) asm volatile("s_wait_asynccnt " #n)
#endif

// ---------------------------------------------------------------------------
// Elementwise f32 -> bf16 conversion (vectorized x4)
// ---------------------------------------------------------------------------
__global__ __launch_bounds__(256)
void cvt_f32_bf16(const float* __restrict__ in, unsigned short* __restrict__ out, int n4) {
    int i = blockIdx.x * blockDim.x + threadIdx.x;
    if (i >= n4) return;
    float4 v = ((const float4*)in)[i];
    uint2 p;
    p.x = (unsigned)f32_to_bf16(v.x) | ((unsigned)f32_to_bf16(v.y) << 16);
    p.y = (unsigned)f32_to_bf16(v.z) | ((unsigned)f32_to_bf16(v.w) << 16);
    ((uint2*)out)[i] = p;
}

// ---------------------------------------------------------------------------
// One-time weight transpose + convert: W[K,N] f32 -> Wt[N,K] bf16.
// 32x32 tiles through LDS; both global accesses coalesced.
// ---------------------------------------------------------------------------
__global__ __launch_bounds__(256)
void transpose_cvt_bf16(const float* __restrict__ W, unsigned short* __restrict__ Wt,
                        int K, int N) {
    __shared__ float t[32][33];
    int kb = blockIdx.y * 32, nb = blockIdx.x * 32;
    int tx = threadIdx.x & 31, ty = threadIdx.x >> 5;   // 32 x 8
    #pragma unroll
    for (int i = 0; i < 32; i += 8)
        t[ty + i][tx] = W[(size_t)(kb + ty + i) * N + (nb + tx)];
    __syncthreads();
    #pragma unroll
    for (int i = 0; i < 32; i += 8)
        Wt[(size_t)(nb + ty + i) * K + (kb + tx)] = f32_to_bf16(t[tx][ty + i]);
}

// ---------------------------------------------------------------------------
// Wide part: row-sum of [B, 256] -> [B]. One wave32 per row, 8 waves/block.
// ---------------------------------------------------------------------------
__global__ __launch_bounds__(256)
void wide_sum_kernel(const float* __restrict__ wf, float* __restrict__ out) {
    int row  = blockIdx.x * 8 + (threadIdx.x >> 5);
    int lane = threadIdx.x & 31;
    const float* r = wf + (size_t)row * 256;
    float s = 0.f;
    #pragma unroll
    for (int c = 0; c < 8; ++c) s += r[lane + c * 32];
    #pragma unroll
    for (int off = 16; off; off >>= 1) s += __shfl_xor(s, off, 32);
    if (lane == 0) out[row] = s;
}

// ---------------------------------------------------------------------------
// GEMM: C = act(A[M,K](bf16,row-major) @ Bt[N,K](bf16,row-major)^T + bias[N])
//   - 256 threads = 8 waves; block tile 256(M) x 64(N); K-step 32
//   - each wave: 32 rows x 64 cols = 8x v_wmma_f32_16x16x32_bf16 per K-step
//   - async global->LDS double buffering (ASYNCcnt) when available,
//     else register double-buffering over a single LDS stage
// LDS fragment layout per ISA 7.12.2 (16-bit A 16x32):
//   lanes 0-15  : row = lane,    K = {0..7, 16..23}
//   lanes 16-31 : row = lane-16, K = {8..15, 24..31}
// Bt rows are output columns, so B fragments use identical address math.
// ---------------------------------------------------------------------------
#define TILE_M 256
#define TILE_N 64
#define TILE_K 32
#define LDSS   40   // 32 + 8 pad: 80B rows, 16B aligned, conflict-free b128

#if HAVE_ASYNC_LDS
#define NBUF 2
#else
#define NBUF 1
#endif

__device__ __forceinline__ v16bf load_frag(const unsigned short* p) {
    v8bf lo = *(const v8bf*)p;          // K = hi*8 .. +7
    v8bf hi = *(const v8bf*)(p + 16);   // K = 16 + hi*8 .. +7
    return __builtin_shufflevector(lo, hi,
        0, 1, 2, 3, 4, 5, 6, 7, 8, 9, 10, 11, 12, 13, 14, 15);
}

__global__ __launch_bounds__(256)
void gemm_bias_relu_bf16(const unsigned short* __restrict__ A,
                         const unsigned short* __restrict__ Bt,
                         const float* __restrict__ bias,
                         float* __restrict__ outF,
                         unsigned short* __restrict__ outB,
                         int M, int N, int K, int relu) {
    __shared__ __align__(16) unsigned short lA[NBUF][TILE_M * LDSS];
    __shared__ __align__(16) unsigned short lB[NBUF][TILE_N * LDSS];

    const int tid  = threadIdx.x;
    const int wave = tid >> 5;
    const int lane = tid & 31;
    const int hi   = lane >> 4;
    const int lr   = lane & 15;

    const int m0 = blockIdx.y * TILE_M;
    const int n0 = blockIdx.x * TILE_N;
    const int mr = wave * 32;           // wave's 32-row slice of block tile

    v8f acc[2][4];
    #pragma unroll
    for (int s = 0; s < 2; ++s)
        #pragma unroll
        for (int t = 0; t < 4; ++t) acc[s][t] = (v8f)(0.0f);

    // Per-thread staging chunks: A = 4 x 16B (1024 chunks), B = 1 x 16B (256)
    const int ar[4] = { (tid + 0*256) >> 2, (tid + 1*256) >> 2,
                        (tid + 2*256) >> 2, (tid + 3*256) >> 2 };
    const int ac    = (tid & 3) * 8;
    const int br    = tid >> 2;
    const int bc    = (tid & 3) * 8;

#if HAVE_ASYNC_LDS
    auto stage = [&](int buf, int k0) {
        #pragma unroll
        for (int i = 0; i < 4; ++i)
            GL_ASYNC_B128(A + (size_t)(m0 + ar[i]) * K + (k0 + ac),
                          &lA[buf][ar[i] * LDSS + ac]);
        GL_ASYNC_B128(Bt + (size_t)(n0 + br) * K + (k0 + bc),
                      &lB[buf][br * LDSS + bc]);
    };
#else
    uint4 rga[4], rgb;
    auto gload = [&](int k0) {
        #pragma unroll
        for (int i = 0; i < 4; ++i)
            rga[i] = *(const uint4*)(A + (size_t)(m0 + ar[i]) * K + (k0 + ac));
        rgb = *(const uint4*)(Bt + (size_t)(n0 + br) * K + (k0 + bc));
    };
    auto lstore = [&]() {
        #pragma unroll
        for (int i = 0; i < 4; ++i)
            *(uint4*)(&lA[0][ar[i] * LDSS + ac]) = rga[i];
        *(uint4*)(&lB[0][br * LDSS + bc]) = rgb;
    };
#endif

    auto compute = [&](int buf) {
        v16bf bfrag[4];
        #pragma unroll
        for (int t = 0; t < 4; ++t)
            bfrag[t] = load_frag(&lB[buf][(t * 16 + lr) * LDSS + hi * 8]);
        #pragma unroll
        for (int s = 0; s < 2; ++s) {
            v16bf afrag = load_frag(&lA[buf][(mr + s * 16 + lr) * LDSS + hi * 8]);
            #pragma unroll
            for (int t = 0; t < 4; ++t)
                acc[s][t] = __builtin_amdgcn_wmma_f32_16x16x32_bf16(
                    false, afrag, false, bfrag[t],
                    (short)0, acc[s][t], false, false);
        }
    };

#if HAVE_ASYNC_LDS
    int buf = 0;
    stage(0, 0);                               // 5 async ops in flight
    for (int k0 = 0; k0 < K; k0 += TILE_K) {
        if (k0 + TILE_K < K) {
            stage(buf ^ 1, k0 + TILE_K);       // +5 for next tile
            WAIT_ASYNC(5);                     // retire current tile (in-order)
        } else {
            WAIT_ASYNC(0);
        }
        __syncthreads();                       // LDS visible to all waves
        compute(buf);
        __syncthreads();                       // done reading before reuse
        buf ^= 1;
    }
#else
    gload(0);
    lstore();
    __syncthreads();
    for (int k0 = 0; k0 < K; k0 += TILE_K) {
        bool next = (k0 + TILE_K) < K;
        if (next) gload(k0 + TILE_K);          // global loads fly over compute
        compute(0);
        __syncthreads();
        if (next) { lstore(); }
        __syncthreads();
    }
#endif

    // ---- epilogue: bias + ReLU, dual-format store ----
    // c[j] -> row = mr + s*16 + hi*8 + j, col = t*16 + lr
    #pragma unroll
    for (int s = 0; s < 2; ++s) {
        #pragma unroll
        for (int t = 0; t < 4; ++t) {
            int col = n0 + t * 16 + lr;
            float bv = bias[col];
            #pragma unroll
            for (int j = 0; j < 8; ++j) {
                int row = m0 + mr + s * 16 + hi * 8 + j;
                float v = acc[s][t][j] + bv;
                if (relu) v = fmaxf(v, 0.f);
                size_t o = (size_t)row * N + col;
                if (outF) outF[o] = v;
                if (outB) outB[o] = f32_to_bf16(v);
            }
        }
    }
}

// ---------------------------------------------------------------------------
// Per-sample domain head: out = sigmoid(wide + dot(d, Wh[dom]) + bh[dom])
// One wave32 per sample.
// ---------------------------------------------------------------------------
__global__ __launch_bounds__(256)
void head_kernel(const float* __restrict__ d,      // [B,256]
                 const int* __restrict__ dom,      // [B]
                 const float* __restrict__ Wh,     // [8,256]
                 const float* __restrict__ bh,     // [8]
                 const float* __restrict__ wide,   // [B]
                 float* __restrict__ out) {        // [B]
    int row  = blockIdx.x * 8 + (threadIdx.x >> 5);
    int lane = threadIdx.x & 31;
    int dm   = dom[row];
    const float* dr = d  + (size_t)row * 256;
    const float* wr = Wh + (size_t)dm  * 256;
    float s = 0.f;
    #pragma unroll
    for (int c = 0; c < 8; ++c) {
        int i = lane + c * 32;
        s += dr[i] * wr[i];
    }
    #pragma unroll
    for (int off = 16; off; off >>= 1) s += __shfl_xor(s, off, 32);
    if (lane == 0) {
        float x = wide[row] + s + bh[dm];
        out[row] = 1.0f / (1.0f + __expf(-x));
    }
}

// ---------------------------------------------------------------------------
// Launch
// ---------------------------------------------------------------------------
extern "C" void kernel_launch(void* const* d_in, const int* in_sizes, int n_in,
                              void* d_out, int out_size, void* d_ws, size_t ws_size,
                              hipStream_t stream) {
    (void)in_sizes; (void)n_in; (void)out_size; (void)ws_size;

    const int   Bn = 16384;
    const float* wide_f = (const float*)d_in[0];   // [B,256]
    const float* deep_f = (const float*)d_in[1];   // [B,1024]
    const int*   dom    = (const int*)d_in[2];     // [B]
    const float* W1  = (const float*)d_in[3];  const float* b1  = (const float*)d_in[4];
    const float* W2  = (const float*)d_in[5];  const float* b2  = (const float*)d_in[6];
    const float* W3  = (const float*)d_in[7];  const float* b3  = (const float*)d_in[8];
    const float* Wd1 = (const float*)d_in[9];  const float* bd1 = (const float*)d_in[10];
    const float* Wd2 = (const float*)d_in[11]; const float* bd2 = (const float*)d_in[12];
    const float* Wh  = (const float*)d_in[13]; const float* bh  = (const float*)d_in[14];

    // Output layout: (out [B], shared [B,512], d [B,256]) concatenated
    float* outP    = (float*)d_out;
    float* sharedP = outP + Bn;
    float* dP      = sharedP + (size_t)Bn * 512;

    // Workspace carve-out (16B-aligned slices)
    char* ws = (char*)d_ws;
    size_t off = 0;
    float* wideS = (float*)(ws + off);                  off += (size_t)Bn * 4;
    unsigned short* Xb   = (unsigned short*)(ws + off); off += (size_t)Bn * 1024 * 2;
    unsigned short* H1b  = (unsigned short*)(ws + off); off += (size_t)Bn * 2048 * 2;
    unsigned short* H2b  = (unsigned short*)(ws + off); off += (size_t)Bn * 1024 * 2;
    unsigned short* Sb   = (unsigned short*)(ws + off); off += (size_t)Bn * 512 * 2;
    unsigned short* D1b  = (unsigned short*)(ws + off); off += (size_t)Bn * 512 * 2;
    unsigned short* W1t  = (unsigned short*)(ws + off); off += (size_t)1024 * 2048 * 2;
    unsigned short* W2t  = (unsigned short*)(ws + off); off += (size_t)2048 * 1024 * 2;
    unsigned short* W3t  = (unsigned short*)(ws + off); off += (size_t)1024 * 512 * 2;
    unsigned short* Wd1t = (unsigned short*)(ws + off); off += (size_t)512 * 512 * 2;
    unsigned short* Wd2t = (unsigned short*)(ws + off); off += (size_t)512 * 256 * 2;

    // deep activations -> bf16 (elementwise)
    {
        int n4 = (Bn * 1024) / 4;
        cvt_f32_bf16<<<(n4 + 255) / 256, 256, 0, stream>>>(deep_f, Xb, n4);
    }
    // weights -> transposed bf16 Wt[N,K] (one-time, L2-resident afterwards)
    auto tw = [&](const float* W, unsigned short* Wt, int K, int N) {
        dim3 grid(N / 32, K / 32);
        transpose_cvt_bf16<<<grid, 256, 0, stream>>>(W, Wt, K, N);
    };
    tw(W1,  W1t,  1024, 2048);
    tw(W2,  W2t,  2048, 1024);
    tw(W3,  W3t,  1024, 512);
    tw(Wd1, Wd1t, 512,  512);
    tw(Wd2, Wd2t, 512,  256);

    // Wide row-sum
    wide_sum_kernel<<<Bn / 8, 256, 0, stream>>>(wide_f, wideS);

    auto gemm = [&](const unsigned short* A, const unsigned short* Bt,
                    const float* bias, float* oF, unsigned short* oB,
                    int N, int K) {
        dim3 grid(N / TILE_N, Bn / TILE_M);
        gemm_bias_relu_bf16<<<grid, 256, 0, stream>>>(A, Bt, bias, oF, oB, Bn, N, K, 1);
    };

    // Shared MLP: 1024 -> 2048 -> 1024 -> 512
    gemm(Xb,  W1t, b1, nullptr, H1b, 2048, 1024);
    gemm(H1b, W2t, b2, nullptr, H2b, 1024, 2048);
    gemm(H2b, W3t, b3, sharedP, Sb,  512,  1024);   // 'shared' output + bf16 feed
    // Domain tower: 512 -> 512 -> 256
    gemm(Sb,  Wd1t, bd1, nullptr, D1b, 512, 512);
    gemm(D1b, Wd2t, bd2, dP, nullptr, 256, 512);    // 'd' output (f32, read by head)

    // Routed head + sigmoid
    head_kernel<<<Bn / 8, 256, 0, stream>>>(dP, dom, Wh, bh, wideS, outP);
}
